// EdgePredictor_16174846837134
// MI455X (gfx1250) — compile-verified, matrix operand order checked by reading, not run on hardware
//
#include <hip/hip_runtime.h>
#include <hip/hip_bf16.h>

typedef __attribute__((ext_vector_type(16))) __bf16 v16bf;
typedef __attribute__((ext_vector_type(8)))  __bf16 v8bf;
typedef __attribute__((ext_vector_type(4)))  __bf16 v4bf;
typedef __attribute__((ext_vector_type(8)))  float  v8f;
typedef __attribute__((ext_vector_type(4)))  float  v4f;

#define NODE_DIM 128
#define EDGE_DIM 16
#define HIDDEN   128
#define N2       64
#define K1REAL   272            // 2*NODE_DIM + EDGE_DIM
#define K1PAD    288            // padded to 9 * 32
#define WAVES    4
#define EPW      16             // edges per wave (one 16-row M tile)
#define EPB      (WAVES * EPW)  // 64 edges per tile-iteration
#define TILES    4              // tile-iterations per block (amortize weights)
#define EPBLK    (EPB * TILES)  // 256 edges per block

// One 16-bit WMMA operand fragment from LDS: two contiguous 8-element bf16
// chunks at K and K+16 -> two ds_load_b128 per lane.
__device__ __forceinline__ v16bf ld_frag_lds(const __bf16* p) {
  v8bf lo = *reinterpret_cast<const v8bf*>(p);
  v8bf hg = *reinterpret_cast<const v8bf*>(p + 16);
  v16bf r;
#pragma unroll
  for (int i = 0; i < 8; ++i) { r[i] = lo[i]; r[8 + i] = hg[i]; }
  return r;
}

__device__ __forceinline__ v4bf cvt4(v4f v) {
  v4bf o;
#pragma unroll
  for (int j = 0; j < 4; ++j) o[j] = (__bf16)v[j];
  return o;
}

__global__ __launch_bounds__(128) void edge_mlp_wmma_kernel(
    const float* __restrict__ x,   const int* __restrict__ ei,
    const float* __restrict__ ea,  const float* __restrict__ W1,
    const float* __restrict__ b1,  const float* __restrict__ W2,
    const float* __restrict__ b2,  const float* __restrict__ W3,
    const float* __restrict__ b3,  float* __restrict__ out, int E)
{
  __shared__ __align__(16) __bf16 sW1[HIDDEN * K1PAD];       // W1^T [n][k] bf16
  __shared__ __align__(16) __bf16 sW2[N2 * HIDDEN];          // W2^T [n][k] bf16
  __shared__ float sB1[HIDDEN];
  __shared__ float sB2[N2];
  __shared__ float sW3[N2];
  __shared__ __align__(16) __bf16 sA [WAVES][EPW * K1PAD];   // edge inputs
  __shared__ __align__(16) __bf16 sH1[WAVES][EPW * HIDDEN];  // layer-1 act
  __shared__ __align__(16) __bf16 sH2[WAVES][EPW * N2];      // layer-2 act

  const int tid  = threadIdx.x;
  const int wave = tid >> 5;
  const int lane = tid & 31;
  const int nl   = lane & 15;      // M (A) / N (B) index within 16-wide tile
  const int half = lane >> 4;      // 0: k 0-7,16-23; 1: k 8-15,24-31 of each 32-step
  const int koff = half * 8;

  // ---- stage weights once per block: 4x4 micro-tile transpose, fp32->bf16 ----
  // W1: [272][128] row-major -> sW1 = W1^T [128][288], K padded with zeros.
  for (int i = tid; i < (K1REAL / 4) * (HIDDEN / 4); i += 128) {
    int k4 = i / (HIDDEN / 4), n4 = i - k4 * (HIDDEN / 4);
    v4f w[4];
#pragma unroll
    for (int j = 0; j < 4; ++j)
      w[j] = *reinterpret_cast<const v4f*>(W1 + (k4 * 4 + j) * HIDDEN + n4 * 4);
#pragma unroll
    for (int j = 0; j < 4; ++j) {
      v4bf o;
#pragma unroll
      for (int m = 0; m < 4; ++m) o[m] = (__bf16)w[m][j];
      *reinterpret_cast<v4bf*>(&sW1[(n4 * 4 + j) * K1PAD + k4 * 4]) = o;
    }
  }
  if (tid < HIDDEN) {             // zero W1 K padding 272..287 for each n
    v8bf z = {};
    *reinterpret_cast<v8bf*>(&sW1[tid * K1PAD + K1REAL])     = z;
    *reinterpret_cast<v8bf*>(&sW1[tid * K1PAD + K1REAL + 8]) = z;
  }
  // A-tile K padding never changes: zero it once (16 rows per wave).
  {
    v8bf z = {};
    if (lane < EPW) {
      *reinterpret_cast<v8bf*>(&sA[wave][lane * K1PAD + K1REAL])     = z;
      *reinterpret_cast<v8bf*>(&sA[wave][lane * K1PAD + K1REAL + 8]) = z;
    }
  }
  // W2: [128][64] -> sW2 = W2^T [64][128]
  for (int i = tid; i < (HIDDEN / 4) * (N2 / 4); i += 128) {
    int k4 = i / (N2 / 4), n4 = i - k4 * (N2 / 4);
    v4f w[4];
#pragma unroll
    for (int j = 0; j < 4; ++j)
      w[j] = *reinterpret_cast<const v4f*>(W2 + (k4 * 4 + j) * N2 + n4 * 4);
#pragma unroll
    for (int j = 0; j < 4; ++j) {
      v4bf o;
#pragma unroll
      for (int m = 0; m < 4; ++m) o[m] = (__bf16)w[m][j];
      *reinterpret_cast<v4bf*>(&sW2[(n4 * 4 + j) * HIDDEN + k4 * 4]) = o;
    }
  }
  if (tid < HIDDEN) sB1[tid] = b1[tid];
  if (tid < N2)     { sB2[tid] = b2[tid]; sW3[tid] = W3[tid]; }
  __syncthreads();

  const float b3v = b3[0];
  const int blockBase = blockIdx.x * EPBLK;

  for (int t = 0; t < TILES; ++t) {
    const int e0 = blockBase + t * EPB + wave * EPW;

    // ---- gather [x[src] | x[dst] | edge_attr] -> bf16 LDS (branch-light) ----
    for (int r = 0; r < EPW; ++r) {
      int e = e0 + r;
      int s = ei[e];
      int d = ei[E + e];
      __bf16* rowp = &sA[wave][r * K1PAD];
      v4f vs = reinterpret_cast<const v4f*>(x + s * NODE_DIM)[lane];
      v4f vd = reinterpret_cast<const v4f*>(x + d * NODE_DIM)[lane];
      *reinterpret_cast<v4bf*>(rowp + lane * 4)            = cvt4(vs);
      *reinterpret_cast<v4bf*>(rowp + NODE_DIM + lane * 4) = cvt4(vd);
      if (lane < EDGE_DIM / 4) {
        v4f ve = reinterpret_cast<const v4f*>(ea + e * EDGE_DIM)[lane];
        *reinterpret_cast<v4bf*>(rowp + 2 * NODE_DIM + lane * 4) = cvt4(ve);
      }
    }
    __syncthreads();

    // ---- layer 1: [16 x 288] @ [288 x 128], 9 K-steps x 8 N-tiles WMMA ----
    v8f acc1[8] = {};
    const __bf16* aBase1 = &sA[wave][nl * K1PAD + koff];
#pragma unroll
    for (int kk = 0; kk < K1PAD / 32; ++kk) {
      v16bf a = ld_frag_lds(aBase1 + kk * 32);
      v16bf bfr[8];
#pragma unroll
      for (int nt = 0; nt < 8; ++nt)
        bfr[nt] = ld_frag_lds(&sW1[(nt * 16 + nl) * K1PAD + kk * 32 + koff]);
#pragma unroll
      for (int nt = 0; nt < 8; ++nt)
        acc1[nt] = __builtin_amdgcn_wmma_f32_16x16x32_bf16(
            false, a, false, bfr[nt], (short)0, acc1[nt], false, false);
    }

    // bias + relu -> sH1 (C layout: VGPR j -> row M = j + 8*half, col = nl)
#pragma unroll
    for (int nt = 0; nt < 8; ++nt) {
      int n = nt * 16 + nl;
      float bias = sB1[n];
#pragma unroll
      for (int j = 0; j < 8; ++j) {
        int M = j + half * 8;
        sH1[wave][M * HIDDEN + n] = (__bf16)fmaxf(acc1[nt][j] + bias, 0.0f);
      }
    }
    __syncthreads();

    // ---- layer 2: [16 x 128] @ [128 x 64], 4 K-steps x 4 N-tiles WMMA ----
    v8f acc2[4] = {};
    const __bf16* aBase2 = &sH1[wave][nl * HIDDEN + koff];
#pragma unroll
    for (int kk = 0; kk < HIDDEN / 32; ++kk) {
      v16bf a = ld_frag_lds(aBase2 + kk * 32);
      v16bf bfr[4];
#pragma unroll
      for (int nt = 0; nt < 4; ++nt)
        bfr[nt] = ld_frag_lds(&sW2[(nt * 16 + nl) * HIDDEN + kk * 32 + koff]);
#pragma unroll
      for (int nt = 0; nt < 4; ++nt)
        acc2[nt] = __builtin_amdgcn_wmma_f32_16x16x32_bf16(
            false, a, false, bfr[nt], (short)0, acc2[nt], false, false);
    }

#pragma unroll
    for (int nt = 0; nt < 4; ++nt) {
      int n = nt * 16 + nl;
      float bias = sB2[n];
#pragma unroll
      for (int j = 0; j < 8; ++j) {
        int M = j + half * 8;
        sH2[wave][M * N2 + n] = (__bf16)fmaxf(acc2[nt][j] + bias, 0.0f);
      }
    }
    __syncthreads();

    // ---- layer 3: 64 -> 1 dot + tanh (one edge per lane, lanes 0..15) ----
    if (lane < EPW) {
      float sum = b3v;
      const v8bf* h2p = reinterpret_cast<const v8bf*>(&sH2[wave][lane * N2]);
#pragma unroll
      for (int c = 0; c < N2 / 8; ++c) {
        v8bf h = h2p[c];
#pragma unroll
        for (int j = 0; j < 8; ++j)
          sum += (float)h[j] * sW3[c * 8 + j];
      }
      out[e0 + lane] = tanhf(sum);
    }
    __syncthreads();
  }
}

extern "C" void kernel_launch(void* const* d_in, const int* in_sizes, int n_in,
                              void* d_out, int out_size, void* d_ws, size_t ws_size,
                              hipStream_t stream) {
  (void)n_in; (void)d_ws; (void)ws_size; (void)out_size;
  const float* x  = (const float*)d_in[0];
  const int*   ei = (const int*)  d_in[1];
  const float* ea = (const float*)d_in[2];
  const float* W1 = (const float*)d_in[3];
  const float* b1 = (const float*)d_in[4];
  const float* W2 = (const float*)d_in[5];
  const float* b2 = (const float*)d_in[6];
  const float* W3 = (const float*)d_in[7];
  const float* b3 = (const float*)d_in[8];
  float* out = (float*)d_out;

  const int E = in_sizes[2] / EDGE_DIM;   // 640000
  const int blocks = E / EPBLK;           // 2500
  hipLaunchKernelGGL(edge_mlp_wmma_kernel, dim3(blocks), dim3(128), 0, stream,
                     x, ei, ea, W1, b1, W2, b2, W3, b3, out, E);
}
